// NADE_54906861912229
// MI455X (gfx1250) — compile-verified, matrix operand order checked by reading, not run on hardware
//
#include <hip/hip_runtime.h>
#include <hip/hip_bf16.h>

typedef __attribute__((ext_vector_type(16))) _Float16 v16h;
typedef __attribute__((ext_vector_type(8)))  float    v8f;

#define D_DIM 784
#define H_DIM 2048
#define B_DIM 4096
#define ROWS_PER_WG   16
#define WAVES_PER_WG  16
#define THREADS_PER_WG (WAVES_PER_WG * 32)
#define COLS_PER_WAVE (H_DIM / WAVES_PER_WG)   // 128
#define TILES_PER_WAVE (COLS_PER_WAVE / 16)    // 8

__device__ __forceinline__ float tanhf_fast(float x) {
#if __has_builtin(__builtin_amdgcn_tanhf)
    return __builtin_amdgcn_tanhf(x);          // v_tanh_f32 (confirmed)
#else
    float e = __builtin_amdgcn_exp2f(-2.885390081777927f * x);  // e^(-2x)
    return fmaf(2.0f, __builtin_amdgcn_rcpf(1.0f + e), -1.0f);
#endif
}

__device__ __forceinline__ float sigmoidf_fast(float x) {
    // sigmoid(x) = 0.5*tanh(x/2) + 0.5. The trailing FMA is deliberately kept
    // in the per-element path: it is the independent VALU op that fills the
    // TRANS-op hazard slot after v_tanh_f32 (removing it costs v_nops).
    return fmaf(0.5f, tanhf_fast(x * 0.5f), 0.5f);
}

// One-shot transpose: Wp[H_DIM][D_DIM] -> Wt[D_DIM][H_DIM] so the scan reads
// contiguous rows (W column gather stride-784 would thrash caches 784 times).
__global__ __launch_bounds__(256)
void nade_transpose_w(const float* __restrict__ Wp, float* __restrict__ Wt) {
    __shared__ float tile[32][33];
    const int dBase = blockIdx.x * 32;
    const int hBase = blockIdx.y * 32;
    for (int i = threadIdx.y; i < 32; i += 8) {
        int h = hBase + i;                // always < H_DIM (2048 % 32 == 0)
        int d = dBase + threadIdx.x;
        tile[i][threadIdx.x] = (d < D_DIM) ? Wp[(size_t)h * D_DIM + d] : 0.0f;
    }
    __syncthreads();
    for (int i = threadIdx.y; i < 32; i += 8) {
        int d = dBase + i;
        int h = hBase + threadIdx.x;
        if (d < D_DIM) Wt[(size_t)d * H_DIM + h] = tile[threadIdx.x][i];
    }
}

// Sequential NADE scan. WG = 16 waves, 16 batch rows; wave w owns H columns
// [w*128, w*128+128). State a[16][128] is register-resident in WMMA C layout
// (8 x v8f per lane). Rank-1 update a += x (outer) W_d runs on the matrix pipe
// via v_wmma_f32_16x16x32_f16 (A,B have only K=0 nonzero), co-executing with
// the VALU sigmoid/dot stream.
__global__ __launch_bounds__(THREADS_PER_WG)
void nade_scan(const float* __restrict__ xf,   // [B, D]
               const float* __restrict__ V,    // [D, H]
               const float* __restrict__ b,    // [D]
               const float* __restrict__ Wsrc, // W_d[j] = Wsrc[d*wd + j*wc]
               long wd, long wc,
               const float* __restrict__ c,    // [H]
               float* __restrict__ out)        // [B, D]
{
    __shared__ float acc[2][ROWS_PER_WG];

    const int tid    = threadIdx.x;
    const int lane   = tid & 31;
    const int wave   = tid >> 5;
    const int lane16 = lane & 15;
    const int half   = lane >> 4;
    const bool lo16  = (half == 0);
    const int colbase = wave * COLS_PER_WAVE;
    const int rowbase = blockIdx.x * ROWS_PER_WG;

    if (tid < 2 * ROWS_PER_WG) ((float*)acc)[tid] = 0.0f;

    // a0 = broadcast of c along batch rows (C layout: VGPR v holds row v+8*half)
    v8f aC[TILES_PER_WAVE];
#pragma unroll
    for (int t = 0; t < TILES_PER_WAVE; ++t) {
        float cv = c[colbase + t * 16 + lane16];
        aC[t] = (v8f){cv, cv, cv, cv, cv, cv, cv, cv};
    }

    // WMMA operands built ONCE: the zero elements stay resident in registers
    // across the whole scan; only element 0 (the K=0 slot) is rewritten.
    // 16-bit A layout: VGPR0 low half = K=0 (lanes 0-15) / K=8 (lanes 16-31);
    // 16-bit B layout: VGPR0 low half = (K=0, N=lane16) / (K=8, ...). Lanes
    // 16-31 get 0.0 via select, so only K=0 contributes to the product.
    //
    // FOUR rotating B buffers: the write to Bm[t&3] lands ~7 instructions
    // after the WMMA that last read it, clearing the 4-slot WMMA->VALU WAR
    // hazard window so the compiler emits (almost) no v_nop padding.
    v16h Amat = {};
    v16h Bm[4] = {{}, {}, {}, {}};

    // Per-lane streaming pointers (advance by one row per step).
    const float* vp = V + colbase + lane16;                      // += H_DIM
    const float* wp = Wsrc + (size_t)(colbase + lane16) * wc;    // += wd
    const float* xp = xf + (size_t)(rowbase + lane16) * D_DIM;   // += 1

    __syncthreads();

    for (int d = 0; d < D_DIM; ++d) {
        const int par = d & 1;

        // Branch-free batched loads (lanes 16-31 mirror lanes 0-15 addresses).
        float vv[TILES_PER_WAVE];
        float wvr[TILES_PER_WAVE];
#pragma unroll
        for (int t = 0; t < TILES_PER_WAVE; ++t) vv[t] = vp[t * 16];
#pragma unroll
        for (int t = 0; t < TILES_PER_WAVE; ++t)
            wvr[t] = wp[(size_t)(t * 16) * wc];
        float xv = *xp;
        xv = lo16 ? xv : 0.0f;                 // v_cndmask, no exec branch
        Amat[0] = (_Float16)xv;

        float dotp[8];
#pragma unroll
        for (int v = 0; v < 8; ++v) dotp[v] = 0.0f;

#pragma unroll
        for (int t = 0; t < TILES_PER_WAVE; ++t) {
            // h = sigmoid(a) (pre-update) ; dot partial with V_d
#pragma unroll
            for (int v = 0; v < 8; ++v) {
                float h = sigmoidf_fast(aC[t][v]);
                dotp[v] = fmaf(h, vv[t], dotp[v]);
            }
            // a_tile += x (outer) w  — matrix pipe, f32 accumulate.
            float wv = lo16 ? wvr[t] : 0.0f;   // v_cndmask
            Bm[t & 3][0] = (_Float16)wv;
            aC[t] = __builtin_amdgcn_wmma_f32_16x16x32_f16(
                        false, Amat, false, Bm[t & 3], (short)0, aC[t],
                        false, false);
        }

        // Prefetch next step's V / W rows (global_prefetch_b8).
        if (d + 1 < D_DIM) {
            __builtin_prefetch(vp + H_DIM, 0, 3);
            __builtin_prefetch(wp + wd, 0, 3);
        }
        vp += H_DIM;
        wp += wd;
        xp += 1;

        // Reduce dot partials across the 16 lanes of each half-wave.
#pragma unroll
        for (int v = 0; v < 8; ++v) {
            float s = dotp[v];
            s += __shfl_xor(s, 1, 32);
            s += __shfl_xor(s, 2, 32);
            s += __shfl_xor(s, 4, 32);
            s += __shfl_xor(s, 8, 32);
            dotp[v] = s;
        }
        if (lane16 == 0) {
#pragma unroll
            for (int v = 0; v < 8; ++v)
                atomicAdd(&acc[par][v + 8 * half], dotp[v]);  // ds_add_f32
        }
        __syncthreads();  // one barrier per step (parity-buffered accumulator)

        if (wave == 0 && lane < ROWS_PER_WG) {
            float p = sigmoidf_fast(acc[par][lane] + b[d]);
            out[(size_t)(rowbase + lane) * D_DIM + d] = p;
            acc[par][lane] = 0.0f;  // recycle for step d+2
        }
    }
}

extern "C" void kernel_launch(void* const* d_in, const int* in_sizes, int n_in,
                              void* d_out, int out_size, void* d_ws, size_t ws_size,
                              hipStream_t stream) {
    const float* x  = (const float*)d_in[0];  // [B, C, H, W] == [B, D]
    const float* V  = (const float*)d_in[1];  // [D, H]
    const float* b  = (const float*)d_in[2];  // [D]
    const float* Wp = (const float*)d_in[3];  // [H, D]
    const float* c  = (const float*)d_in[4];  // [H]
    float* out = (float*)d_out;

    const size_t wt_bytes = (size_t)D_DIM * H_DIM * sizeof(float);
    const float* Wsrc;
    long wd, wc;
    if (ws_size >= wt_bytes) {
        float* Wt = (float*)d_ws;
        nade_transpose_w<<<dim3((D_DIM + 31) / 32, H_DIM / 32), dim3(32, 8), 0,
                           stream>>>(Wp, Wt);
        Wsrc = Wt; wd = H_DIM; wc = 1;          // contiguous rows
    } else {
        Wsrc = Wp; wd = 1;     wc = D_DIM;      // fallback: strided column gather
    }

    nade_scan<<<B_DIM / ROWS_PER_WG, THREADS_PER_WG, 0, stream>>>(
        x, V, b, Wsrc, wd, wc, c, out);
}
